// SSMnet_9182640079547
// MI455X (gfx1250) — matrix operand order repaired and checked by reading, not compiled, not run on hardware
//
#include <hip/hip_runtime.h>
#include <hip/hip_bf16.h>

typedef __attribute__((ext_vector_type(16))) _Float16 v16h;
typedef __attribute__((ext_vector_type(8)))  _Float16 v8h;
typedef __attribute__((ext_vector_type(4)))  _Float16 v4h;
typedef __attribute__((ext_vector_type(8)))  float    v8f;

#define T_SEQ 1024

__device__ __forceinline__ float selu_f(float x) {
    const float a  = 1.6732632423543772f;
    const float sc = 1.0507009873554805f;
    return x > 0.f ? sc * x : sc * a * (__expf(x) - 1.f);
}

__device__ __forceinline__ v4h pack4(float4 f) {
    v4h h;
    h[0] = (_Float16)f.x; h[1] = (_Float16)f.y;
    h[2] = (_Float16)f.z; h[3] = (_Float16)f.w;
    return h;
}

// ---------------------------------------------------------------------------
// Conv1: (T,1,72,64) -> conv6x4 pad(2,3)(1,2) -> GN(per-ch) -> SELU
//        -> maxpool 2x4 -> (T,32,36,16)
// ---------------------------------------------------------------------------
__global__ void __launch_bounds__(256) conv1_kernel(
    const float* __restrict__ x, const float* __restrict__ w,
    const float* __restrict__ b, const float* __restrict__ g,
    const float* __restrict__ gb, float* __restrict__ out)
{
    __shared__ float inb[72 * 64];
    __shared__ float cb[72 * 64];
    __shared__ float red[256], red2[256];
    __shared__ float s_mean, s_rstd;
    const int t = blockIdx.x;
    const int tid = threadIdx.x;
    const float* xin = x + (long)t * 72 * 64;
    for (int i = tid; i < 72 * 64; i += 256) inb[i] = xin[i];
    __syncthreads();

    for (int ch = 0; ch < 32; ++ch) {
        float wr[24];
        #pragma unroll
        for (int i = 0; i < 24; ++i) wr[i] = w[ch * 24 + i];   // uniform -> SGPR
        const float bias = b[ch];
        float lsum = 0.f, lsq = 0.f;
        for (int idx = tid; idx < 4608; idx += 256) {
            int oh = idx >> 6, ow = idx & 63;
            float s = bias;
            #pragma unroll
            for (int kh = 0; kh < 6; ++kh) {
                int ih = oh + kh - 2;
                if ((unsigned)ih >= 72u) continue;
                #pragma unroll
                for (int kw = 0; kw < 4; ++kw) {
                    int iw = ow + kw - 1;
                    if ((unsigned)iw >= 64u) continue;
                    s += wr[kh * 4 + kw] * inb[ih * 64 + iw];
                }
            }
            cb[idx] = s; lsum += s; lsq += s * s;
        }
        red[tid] = lsum; red2[tid] = lsq;
        __syncthreads();
        for (int st = 128; st > 0; st >>= 1) {
            if (tid < st) { red[tid] += red[tid + st]; red2[tid] += red2[tid + st]; }
            __syncthreads();
        }
        if (tid == 0) {
            float m = red[0] * (1.f / 4608.f);
            float v = red2[0] * (1.f / 4608.f) - m * m;
            s_mean = m; s_rstd = rsqrtf(fmaxf(v, 0.f) + 1e-5f);
        }
        __syncthreads();
        const float gm = g[ch], gbb = gb[ch], mean = s_mean, rstd = s_rstd;
        for (int p = tid; p < 576; p += 256) {
            int ph = p >> 4, pw = p & 15;
            float mx = -1e30f;
            #pragma unroll
            for (int dh = 0; dh < 2; ++dh)
                #pragma unroll
                for (int dw = 0; dw < 4; ++dw) {
                    float xv = cb[(ph * 2 + dh) * 64 + (pw * 4 + dw)];
                    xv = (xv - mean) * rstd * gm + gbb;
                    mx = fmaxf(mx, selu_f(xv));
                }
            out[((long)t * 32 + ch) * 576 + p] = mx;
        }
        __syncthreads();
    }
}

// ---------------------------------------------------------------------------
// Conv2: (T,32,36,16) -> conv6x4 -> GN(2ch/grp) -> SELU -> pool 3x4
//        -> (T,64,12,4). Group weights staged in LDS.
// ---------------------------------------------------------------------------
__global__ void __launch_bounds__(256) conv2_kernel(
    const float* __restrict__ in, const float* __restrict__ w,
    const float* __restrict__ b, const float* __restrict__ g,
    const float* __restrict__ gb, float* __restrict__ out)
{
    __shared__ _Float16 inb[32 * 576];
    __shared__ float wbuf[2 * 32 * 24];   // 1536 floats
    __shared__ float cb[2 * 576];
    __shared__ float red[256], red2[256];
    __shared__ float s_mean, s_rstd;
    const int t = blockIdx.x;
    const int tid = threadIdx.x;
    const float* xin = in + (long)t * 32 * 576;
    for (int i = tid; i < 32 * 576; i += 256) inb[i] = (_Float16)xin[i];
    __syncthreads();

    for (int grp = 0; grp < 32; ++grp) {
        for (int i = tid; i < 1536; i += 256) wbuf[i] = w[(long)grp * 1536 + i];
        __syncthreads();
        float lsum = 0.f, lsq = 0.f;
        for (int e = tid; e < 1152; e += 256) {
            int sub = (e >= 576) ? 1 : 0;
            int idx = e - sub * 576;
            int oc = grp * 2 + sub;
            int oh = idx >> 4, ow = idx & 15;
            float s = b[oc];
            const float* wp = wbuf + sub * 768;
            for (int ic = 0; ic < 32; ++ic) {
                const _Float16* ip = inb + ic * 576;
                const float* wq = wp + ic * 24;
                #pragma unroll
                for (int kh = 0; kh < 6; ++kh) {
                    int ih = oh + kh - 2;
                    if ((unsigned)ih >= 36u) continue;
                    #pragma unroll
                    for (int kw = 0; kw < 4; ++kw) {
                        int iw = ow + kw - 1;
                        if ((unsigned)iw >= 16u) continue;
                        s += wq[kh * 4 + kw] * (float)ip[ih * 16 + iw];
                    }
                }
            }
            cb[e] = s; lsum += s; lsq += s * s;
        }
        red[tid] = lsum; red2[tid] = lsq;
        __syncthreads();
        for (int st = 128; st > 0; st >>= 1) {
            if (tid < st) { red[tid] += red[tid + st]; red2[tid] += red2[tid + st]; }
            __syncthreads();
        }
        if (tid == 0) {
            float m = red[0] * (1.f / 1152.f);
            float v = red2[0] * (1.f / 1152.f) - m * m;
            s_mean = m; s_rstd = rsqrtf(fmaxf(v, 0.f) + 1e-5f);
        }
        __syncthreads();
        const float mean = s_mean, rstd = s_rstd;
        for (int p = tid; p < 96; p += 256) {
            int sub = p / 48, pidx = p - sub * 48;
            int oc = grp * 2 + sub;
            int ph = pidx >> 2, pw = pidx & 3;
            const float gm = g[oc], gbb = gb[oc];
            float mx = -1e30f;
            #pragma unroll
            for (int dh = 0; dh < 3; ++dh)
                #pragma unroll
                for (int dw = 0; dw < 4; ++dw) {
                    float xv = cb[sub * 576 + (ph * 3 + dh) * 16 + (pw * 4 + dw)];
                    xv = (xv - mean) * rstd * gm + gbb;
                    mx = fmaxf(mx, selu_f(xv));
                }
            out[((long)t * 64 + oc) * 48 + pidx] = mx;
        }
        __syncthreads();
    }
}

// ---------------------------------------------------------------------------
// Conv3: (T,64,12,4) -> conv6x4 -> GN(4ch/grp) -> SELU -> pool 3x2
//        -> (T,128,4,2) flattened to (T,1024). Group weights staged in LDS.
// ---------------------------------------------------------------------------
__global__ void __launch_bounds__(256) conv3_kernel(
    const float* __restrict__ in, const float* __restrict__ w,
    const float* __restrict__ b, const float* __restrict__ g,
    const float* __restrict__ gb, float* __restrict__ out)
{
    __shared__ float inb[64 * 48];
    __shared__ float wbuf[4 * 64 * 24];   // 6144 floats
    __shared__ float cb[4 * 48];
    __shared__ float red[256], red2[256];
    __shared__ float s_mean, s_rstd;
    const int t = blockIdx.x;
    const int tid = threadIdx.x;
    const float* xin = in + (long)t * 64 * 48;
    for (int i = tid; i < 64 * 48; i += 256) inb[i] = xin[i];
    __syncthreads();

    for (int grp = 0; grp < 32; ++grp) {
        for (int i = tid; i < 6144; i += 256) wbuf[i] = w[(long)grp * 6144 + i];
        __syncthreads();
        float lsum = 0.f, lsq = 0.f;
        if (tid < 192) {
            int sub = tid / 48, idx = tid - sub * 48;
            int oc = grp * 4 + sub;
            int oh = idx >> 2, ow = idx & 3;
            float s = b[oc];
            const float* wp = wbuf + sub * 1536;
            for (int ic = 0; ic < 64; ++ic) {
                const float* ip = inb + ic * 48;
                const float* wq = wp + ic * 24;
                #pragma unroll
                for (int kh = 0; kh < 6; ++kh) {
                    int ih = oh + kh - 2;
                    if ((unsigned)ih >= 12u) continue;
                    #pragma unroll
                    for (int kw = 0; kw < 4; ++kw) {
                        int iw = ow + kw - 1;
                        if ((unsigned)iw >= 4u) continue;
                        s += wq[kh * 4 + kw] * ip[ih * 4 + iw];
                    }
                }
            }
            cb[tid] = s; lsum = s; lsq = s * s;
        }
        red[tid] = lsum; red2[tid] = lsq;
        __syncthreads();
        for (int st = 128; st > 0; st >>= 1) {
            if (tid < st) { red[tid] += red[tid + st]; red2[tid] += red2[tid + st]; }
            __syncthreads();
        }
        if (tid == 0) {
            float m = red[0] * (1.f / 192.f);
            float v = red2[0] * (1.f / 192.f) - m * m;
            s_mean = m; s_rstd = rsqrtf(fmaxf(v, 0.f) + 1e-5f);
        }
        __syncthreads();
        const float mean = s_mean, rstd = s_rstd;
        if (tid < 32) {
            int sub = tid >> 3, pidx = tid & 7;
            int oc = grp * 4 + sub;
            int ph = pidx >> 1, pw = pidx & 1;
            const float gm = g[oc], gbb = gb[oc];
            float mx = -1e30f;
            #pragma unroll
            for (int dh = 0; dh < 3; ++dh)
                #pragma unroll
                for (int dw = 0; dw < 2; ++dw) {
                    float xv = cb[sub * 48 + (ph * 3 + dh) * 4 + (pw * 2 + dw)];
                    xv = (xv - mean) * rstd * gm + gbb;
                    mx = fmaxf(mx, selu_f(xv));
                }
            out[(long)t * 1024 + oc * 8 + pidx] = mx;
        }
        __syncthreads();
    }
}

// ---------------------------------------------------------------------------
// Generic batched WMMA GEMM, f16 LDS staging, f32 accumulate.
// C[m,n] = sum_k A[m,k]*B[k,n]  (transB: B[k,n] = Bmem[n,k])
// Block 128 threads (4 wave32), C tile 64x32, K step 32, LDS double-buffered.
// A staged row-major (stride 40 halves); B staged n-major Bst[n][k] so each
// lane's fragment is two contiguous 16-byte LDS reads (ds_load_b128).
// Global loads are float4 (global_load_b128); LDS stores packed f16 b64.
// epi: 0=none, 1=SELU, 3=0.5+0.5*x. bias added when non-null.
// ---------------------------------------------------------------------------
__global__ void __launch_bounds__(128) gemm_wmma_kernel(
    const float* __restrict__ A, int lda, long aBatch,
    const float* __restrict__ B, int ldb, long bBatch, int transB,
    float* __restrict__ C, int ldc, long cBatch,
    const float* __restrict__ bias,
    int M, int N, int K, int epi)
{
    __shared__ alignas(16) _Float16 As[2][64][40];
    __shared__ alignas(16) _Float16 Bst[2][32][40];
    const int tid  = threadIdx.x;
    const int wave = tid >> 5;
    const int lane = tid & 31;
    const int laneHalf = lane >> 4;
    const int lm = lane & 15;
    const int nBase = blockIdx.x * 32;
    const int mBase = blockIdx.y * 64;
    const int z = blockIdx.z;
    const float* Ab = A + (long)z * aBatch;
    const float* Bb = B + (long)z * bBatch;
    float* Cb = C + (long)z * cBatch;

    v8f acc0 = {}; v8f acc1 = {};
    const int am = wave * 16 + lm;
    const int nk = K >> 5;

    float4 aReg[4];
    float4 bReg[2];

    auto loadGlobal = [&](int kt) {
        const int k0 = kt << 5;
        #pragma unroll
        for (int i = 0; i < 4; ++i) {
            int idx = tid + i * 128;
            int r = idx >> 3, c4 = (idx & 7) << 2;
            aReg[i] = *(const float4*)&Ab[(long)(mBase + r) * lda + (k0 + c4)];
        }
        if (transB) {
            #pragma unroll
            for (int i = 0; i < 2; ++i) {
                int idx = tid + i * 128;
                int n = idx >> 3, c4 = (idx & 7) << 2;
                bReg[i] = *(const float4*)&Bb[(long)(nBase + n) * ldb + (k0 + c4)];
            }
        } else {
            #pragma unroll
            for (int i = 0; i < 2; ++i) {
                int idx = tid + i * 128;
                int k = idx >> 3, n4 = (idx & 7) << 2;
                bReg[i] = *(const float4*)&Bb[(long)(k0 + k) * ldb + (nBase + n4)];
            }
        }
    };

    auto storeLds = [&](int buf) {
        #pragma unroll
        for (int i = 0; i < 4; ++i) {
            int idx = tid + i * 128;
            int r = idx >> 3, c4 = (idx & 7) << 2;
            *(v4h*)&As[buf][r][c4] = pack4(aReg[i]);   // 8B aligned -> ds_store_b64
        }
        if (transB) {
            #pragma unroll
            for (int i = 0; i < 2; ++i) {
                int idx = tid + i * 128;
                int n = idx >> 3, c4 = (idx & 7) << 2;
                *(v4h*)&Bst[buf][n][c4] = pack4(bReg[i]);
            }
        } else {
            #pragma unroll
            for (int i = 0; i < 2; ++i) {
                int idx = tid + i * 128;
                int k = idx >> 3, n4 = (idx & 7) << 2;
                const float* c = &bReg[i].x;
                #pragma unroll
                for (int j = 0; j < 4; ++j)
                    Bst[buf][n4 + j][k] = (_Float16)c[j];
            }
        }
    };

    loadGlobal(0);
    storeLds(0);
    __syncthreads();

    for (int kt = 0; kt < nk; ++kt) {
        const int cur = kt & 1;
        if (kt + 1 < nk) loadGlobal(kt + 1);

        // A fragment: K = laneHalf*16 .. +15, contiguous in As[m][*]
        const v8h* arow = (const v8h*)&As[cur][am][0];       // row = 80B, 16B aligned
        v8h a_lo = arow[laneHalf * 2];
        v8h a_hi = arow[laneHalf * 2 + 1];
        v16h af = __builtin_shufflevector(a_lo, a_hi,
                    0,1,2,3,4,5,6,7,8,9,10,11,12,13,14,15);
        // B fragments: K = laneHalf*16 .. +15, contiguous in Bst[n][*]
        const v8h* brow0 = (const v8h*)&Bst[cur][lm][0];
        v8h b0a = brow0[laneHalf * 2];
        v8h b0b = brow0[laneHalf * 2 + 1];
        v16h bf0 = __builtin_shufflevector(b0a, b0b,
                    0,1,2,3,4,5,6,7,8,9,10,11,12,13,14,15);
        const v8h* brow1 = (const v8h*)&Bst[cur][16 + lm][0];
        v8h b1a = brow1[laneHalf * 2];
        v8h b1b = brow1[laneHalf * 2 + 1];
        v16h bf1 = __builtin_shufflevector(b1a, b1b,
                    0,1,2,3,4,5,6,7,8,9,10,11,12,13,14,15);

        acc0 = __builtin_amdgcn_wmma_f32_16x16x32_f16(false, af, false, bf0,
                                                      (short)0, acc0, false, false);
        acc1 = __builtin_amdgcn_wmma_f32_16x16x32_f16(false, af, false, bf1,
                                                      (short)0, acc1, false, false);

        if (kt + 1 < nk) {
            __syncthreads();
            storeLds(1 - cur);
            __syncthreads();
        }
    }

    const int mrowBase = mBase + wave * 16 + laneHalf * 8;
    #pragma unroll
    for (int r = 0; r < 8; ++r) {
        int m = mrowBase + r;
        int n0 = nBase + lm;
        float val = acc0[r];
        if (bias) val += bias[n0];
        if (epi == 1) val = selu_f(val);
        else if (epi == 3) val = 0.5f + 0.5f * val;
        Cb[(long)m * ldc + n0] = val;

        int n1 = nBase + 16 + lm;
        float val1 = acc1[r];
        if (bias) val1 += bias[n1];
        if (epi == 1) val1 = selu_f(val1);
        else if (epi == 3) val1 = 0.5f + 0.5f * val1;
        Cb[(long)m * ldc + n1] = val1;
    }
}

// ---------------------------------------------------------------------------
// Row L2 normalize: (1024,128)
// ---------------------------------------------------------------------------
__global__ void __launch_bounds__(128) rownorm_kernel(
    const float* __restrict__ in, float* __restrict__ out)
{
    __shared__ float red[128];
    const int t = blockIdx.x, d = threadIdx.x;
    float v = in[(long)t * 128 + d];
    red[d] = v * v;
    __syncthreads();
    for (int st = 64; st > 0; st >>= 1) {
        if (d < st) red[d] += red[d + st];
        __syncthreads();
    }
    float n = fmaxf(sqrtf(red[0]), 1e-12f);
    out[(long)t * 128 + d] = v / n;
}

// ---------------------------------------------------------------------------
// Fused causal softmax with relative-position skew (in place over qk):
// s[j] = (qk[h,i,j] + qer[h,i,1023-i+j]) / sqrt(32) for j<=i, else masked.
// ---------------------------------------------------------------------------
__global__ void __launch_bounds__(256) softmax_kernel(
    const float* __restrict__ qk, const float* __restrict__ qer,
    float* __restrict__ attn)
{
    __shared__ float red[256];
    const int i = blockIdx.x, h = blockIdx.y, tid = threadIdx.x;
    const float scale = 0.1767766952966369f;  // 1/sqrt(32)
    const long base = ((long)h * T_SEQ + i) * T_SEQ;
    float sv[4];
    float lmax = -1e30f;
    #pragma unroll
    for (int c = 0; c < 4; ++c) {
        int j = tid + c * 256;
        float s = -1e30f;
        if (j <= i)
            s = (qk[base + j] + qer[base + (T_SEQ - 1 - i + j)]) * scale;
        sv[c] = s;
        lmax = fmaxf(lmax, s);
    }
    red[tid] = lmax;
    __syncthreads();
    for (int st = 128; st > 0; st >>= 1) {
        if (tid < st) red[tid] = fmaxf(red[tid], red[tid + st]);
        __syncthreads();
    }
    const float rowmax = red[0];
    __syncthreads();
    float lsum = 0.f;
    #pragma unroll
    for (int c = 0; c < 4; ++c) {
        int j = tid + c * 256;
        float p = (j <= i) ? __expf(sv[c] - rowmax) : 0.f;
        sv[c] = p;
        lsum += p;
    }
    red[tid] = lsum;
    __syncthreads();
    for (int st = 128; st > 0; st >>= 1) {
        if (tid < st) red[tid] += red[tid + st];
        __syncthreads();
    }
    const float inv = 1.f / red[0];
    #pragma unroll
    for (int c = 0; c < 4; ++c) {
        int j = tid + c * 256;
        attn[base + j] = sv[c] * inv;
    }
}

// ---------------------------------------------------------------------------
// Column L2 norms over sequence axis, then scale
// ---------------------------------------------------------------------------
__global__ void __launch_bounds__(256) colnorm_kernel(
    const float* __restrict__ in, float* __restrict__ norms)
{
    __shared__ float red[256];
    const int d = blockIdx.x, tid = threadIdx.x;
    float s = 0.f;
    for (int t = tid; t < T_SEQ; t += 256) {
        float v = in[(long)t * 128 + d];
        s += v * v;
    }
    red[tid] = s;
    __syncthreads();
    for (int st = 128; st > 0; st >>= 1) {
        if (tid < st) red[tid] += red[tid + st];
        __syncthreads();
    }
    if (tid == 0) norms[d] = fmaxf(sqrtf(red[0]), 1e-12f);
}

__global__ void __launch_bounds__(256) colscale_kernel(
    const float* __restrict__ in, const float* __restrict__ norms,
    float* __restrict__ out)
{
    int i = blockIdx.x * 256 + threadIdx.x;   // 1024*128 total
    out[i] = in[i] / norms[i & 127];
}

// ---------------------------------------------------------------------------
extern "C" void kernel_launch(void* const* d_in, const int* in_sizes, int n_in,
                              void* d_out, int out_size, void* d_ws, size_t ws_size,
                              hipStream_t stream)
{
    const float* x    = (const float*)d_in[0];
    const float* c1w  = (const float*)d_in[1];
    const float* c1b  = (const float*)d_in[2];
    const float* g1g  = (const float*)d_in[3];
    const float* g1b  = (const float*)d_in[4];
    const float* c2w  = (const float*)d_in[5];
    const float* c2b  = (const float*)d_in[6];
    const float* g2g  = (const float*)d_in[7];
    const float* g2b  = (const float*)d_in[8];
    const float* c3w  = (const float*)d_in[9];
    const float* c3b  = (const float*)d_in[10];
    const float* g3g  = (const float*)d_in[11];
    const float* g3b  = (const float*)d_in[12];
    const float* f1w  = (const float*)d_in[13];
    const float* f1b  = (const float*)d_in[14];
    const float* f2w  = (const float*)d_in[15];
    const float* f2b  = (const float*)d_in[16];
    const float* f3w  = (const float*)d_in[17];
    const float* f3b  = (const float*)d_in[18];
    const float* qw   = (const float*)d_in[19];
    const float* qbias= (const float*)d_in[20];
    const float* kw   = (const float*)d_in[21];
    const float* kbias= (const float*)d_in[22];
    const float* vw   = (const float*)d_in[23];
    const float* vbias= (const float*)d_in[24];
    const float* Er   = (const float*)d_in[25];
    float* outp = (float*)d_out;

    float* p = (float*)d_ws;
    float* buf1 = p; p += (size_t)1024 * 32 * 576;   // conv1 pooled
    float* buf2 = p; p += (size_t)1024 * 64 * 48;    // conv2 pooled
    float* Xb   = p; p += (size_t)1024 * 1024;       // conv3 flattened
    float* f1o  = p; p += (size_t)1024 * 512;
    float* f2o  = p; p += (size_t)1024 * 256;
    float* f3o  = p; p += (size_t)1024 * 128;
    float* eb   = p; p += (size_t)1024 * 128;
    float* qb   = p; p += (size_t)1024 * 128;
    float* kb   = p; p += (size_t)1024 * 128;
    float* vb   = p; p += (size_t)1024 * 128;
    float* qkb  = p; p += (size_t)4 * 1024 * 1024;   // scores -> attn (in place)
    float* qerb = p; p += (size_t)4 * 1024 * 1024;   // QEr
    float* ob   = p; p += (size_t)1024 * 128;        // attn @ v
    float* eab  = p; p += (size_t)1024 * 128;        // column-normalized
    float* nrm  = p; p += 128;

    // CNN (one frame per block, intermediates + group weights LDS-resident)
    conv1_kernel<<<1024, 256, 0, stream>>>(x, c1w, c1b, g1g, g1b, buf1);
    conv2_kernel<<<1024, 256, 0, stream>>>(buf1, c2w, c2b, g2g, g2b, buf2);
    conv3_kernel<<<1024, 256, 0, stream>>>(buf2, c3w, c3b, g3g, g3b, Xb);

    // FC stack (SELU fused)
    gemm_wmma_kernel<<<dim3(16, 16, 1), 128, 0, stream>>>(
        Xb, 1024, 0, f1w, 1024, 0, 1, f1o, 512, 0, f1b, 1024, 512, 1024, 1);
    gemm_wmma_kernel<<<dim3(8, 16, 1), 128, 0, stream>>>(
        f1o, 512, 0, f2w, 512, 0, 1, f2o, 256, 0, f2b, 1024, 256, 512, 1);
    gemm_wmma_kernel<<<dim3(4, 16, 1), 128, 0, stream>>>(
        f2o, 256, 0, f3w, 256, 0, 1, f3o, 128, 0, f3b, 1024, 128, 256, 1);
    rownorm_kernel<<<1024, 128, 0, stream>>>(f3o, eb);

    // Q, K, V projections (bias, no activation)
    gemm_wmma_kernel<<<dim3(4, 16, 1), 128, 0, stream>>>(
        eb, 128, 0, qw, 128, 0, 1, qb, 128, 0, qbias, 1024, 128, 128, 0);
    gemm_wmma_kernel<<<dim3(4, 16, 1), 128, 0, stream>>>(
        eb, 128, 0, kw, 128, 0, 1, kb, 128, 0, kbias, 1024, 128, 128, 0);
    gemm_wmma_kernel<<<dim3(4, 16, 1), 128, 0, stream>>>(
        eb, 128, 0, vw, 128, 0, 1, vb, 128, 0, vbias, 1024, 128, 128, 0);

    // Per-head Q @ K^T  (batched over 4 heads via column offsets)
    gemm_wmma_kernel<<<dim3(32, 16, 4), 128, 0, stream>>>(
        qb, 128, 32, kb, 128, 32, 1, qkb, 1024, (long)1024 * 1024,
        nullptr, 1024, 1024, 32, 0);
    // Per-head Q @ Er^T  (Er rows [1024,2048), shared across heads)
    gemm_wmma_kernel<<<dim3(32, 16, 4), 128, 0, stream>>>(
        qb, 128, 32, Er + (size_t)1024 * 32, 32, 0, 1, qerb, 1024,
        (long)1024 * 1024, nullptr, 1024, 1024, 32, 0);

    // Fused skew + causal mask + scale + softmax (in place over qkb)
    softmax_kernel<<<dim3(1024, 4), 256, 0, stream>>>(qkb, qerb, qkb);

    // attn @ V -> (1024,128) head-concatenated
    gemm_wmma_kernel<<<dim3(1, 16, 4), 128, 0, stream>>>(
        qkb, 1024, (long)1024 * 1024, vb, 128, 32, 0, ob, 128, 32,
        nullptr, 1024, 32, 1024, 0);

    // Normalize over the sequence axis (dim=1 of (1,T,D))
    colnorm_kernel<<<128, 256, 0, stream>>>(ob, nrm);
    colscale_kernel<<<512, 256, 0, stream>>>(ob, nrm, eab);

    // ssm = 0.5 * (1 + ea @ ea^T) -> d_out (1024x1024 f32)
    gemm_wmma_kernel<<<dim3(32, 16, 1), 128, 0, stream>>>(
        eab, 128, 0, eab, 128, 0, 1, outp, 1024, 0,
        nullptr, 1024, 1024, 128, 3);
}
